// GRUCell_11141145166542
// MI455X (gfx1250) — matrix operand order repaired and checked by reading, not compiled, not run on hardware
//
#include <hip/hip_runtime.h>
#include <hip/hip_bf16.h>

// ---------------- problem constants ----------------
#define IN_   1024
#define HIDN  1024
#define TT    128
#define BATCH 256
#define NTB   (TT * BATCH)     // 32768, GEMM N dimension
#define KT32  (HIDN / 32)      // 32 k-tiles of 32 per GEMM
#define HPAD  16               // pad LDS rows: (1024+16)*2B = 2080B, 32B-aligned
#define HROW  (HIDN + HPAD)
#define BROW  48               // gemm LDS B-tile row stride (ushorts): 96B, 32B-aligned
#define BUFE  (128 * BROW)     // one B-tile buffer, in ushorts

typedef __attribute__((ext_vector_type(16))) __bf16 v16bf;
typedef __attribute__((ext_vector_type(8)))  float  v8f;

static __device__ __forceinline__ v8f wmma_bf16(v16bf a, v16bf b, v8f c) {
    // D = A(16x32 bf16) * B(32x16 bf16) + C(16x16 f32)
    return __builtin_amdgcn_wmma_f32_16x16x32_bf16(
        /*neg_a=*/false, a, /*neg_b=*/false, b,
        /*c_mod=*/(short)0, c, /*reuse_a=*/false, /*reuse_b=*/false);
}

static __device__ __forceinline__ unsigned short f2bf(float f) {
    union { float f; unsigned u; } v; v.f = f;
    unsigned r = v.u + 0x7FFFu + ((v.u >> 16) & 1u);   // round-to-nearest-even
    return (unsigned short)(r >> 16);
}
static __device__ __forceinline__ float bf2f(unsigned short h) {
    union { unsigned u; float f; } v; v.u = ((unsigned)h) << 16;
    return v.f;
}

// ---------------------------------------------------------------------------
// Pack a [1024 x 1024] f32 weight matrix into a 16x32 bf16 A-fragment
// layout (contiguous k per lane half, B reads the same order): per tile (mt,kt): lane l (0..31) slot e (0..15):
//   m = mt*16 + (l&15)
//   k = kt*32 + e + ((l>=16) ? 16 : 0)
// Packed element index = ((mt*KT32 + kt)*32 + l)*16 + e  → inner loop loads
// 32 contiguous bytes per lane.
// ---------------------------------------------------------------------------
__global__ void pack_w(const float* __restrict__ w, unsigned short* __restrict__ dst) {
    int idx  = blockIdx.x * 256 + threadIdx.x;      // 1024*1024 total
    int e    = idx & 15;
    int lane = (idx >> 4) & 31;
    int tile = idx >> 9;
    int kt   = tile & (KT32 - 1);
    int mt   = tile >> 5;
    int m    = mt * 16 + (lane & 15);
    int k    = kt * 32 + e + ((lane >> 4) << 4);
    dst[idx] = f2bf(w[(size_t)m * 1024 + k]);
}

// ---------------------------------------------------------------------------
// x [IN, T*B] f32  ->  xT [T*B, IN] bf16  (coalesced both sides via LDS tile)
// ---------------------------------------------------------------------------
__global__ void transpose_x(const float* __restrict__ x, unsigned short* __restrict__ xT) {
    __shared__ float tile[32][33];
    const int n0 = blockIdx.x * 32;      // N (T*B) tile
    const int k0 = blockIdx.y * 32;      // K (IN) tile
    const int tx = threadIdx.x & 31;
    const int ty = threadIdx.x >> 5;     // 0..7
    #pragma unroll
    for (int i = 0; i < 32; i += 8)
        tile[ty + i][tx] = x[(size_t)(k0 + ty + i) * NTB + n0 + tx];
    __syncthreads();
    #pragma unroll
    for (int i = 0; i < 32; i += 8)
        xT[(size_t)(n0 + ty + i) * IN_ + k0 + tx] = f2bf(tile[tx][ty + i]);
}

// ---------------------------------------------------------------------------
// Projection GEMM: p_g[m, n] = sum_k Wg[m,k] * x[k,n]   (M=1024,N=32768,K=1024)
// grid = (N/128, M/128, 3 gates), 256 threads (8 waves), wave w owns 1 M-tile
// row x 8 N-tiles. B tiles (shared by all 8 waves) staged through LDS with
// double buffering; A fragments double-buffered in registers; B fragments
// double-buffered from LDS (ds_load for nt+1 issues before WMMA nt). One
// barrier per k-step; next-tile global loads issue before the current WMMAs.
// ---------------------------------------------------------------------------
__launch_bounds__(256)
__global__ void gemm_proj(const unsigned short* __restrict__ wp,  // packed [3][1024*1024]
                          const unsigned short* __restrict__ xT,  // [NTB][IN] bf16
                          float* __restrict__ p)                  // [3][1024][NTB] f32
{
    __shared__ __attribute__((aligned(32))) unsigned short Bs[2 * BUFE];

    const int tid   = threadIdx.x;
    const int g     = blockIdx.z;
    const int nbase = blockIdx.x * 128;
    const int wave  = tid >> 5;
    const int lane  = tid & 31;
    const int col   = lane & 15;
    const int hi    = lane >> 4;
    const int mt    = blockIdx.y * 8 + wave;        // global 16-row tile index

    const unsigned short* wg = wp + (size_t)g * (1024u * 1024u)
                                  + (size_t)lane * 16;            // + tile*512

    // cooperative B-tile copy: thread -> (seg, half), 32 bytes each
    const int seg  = tid >> 1;            // 0..127 : column within tile
    const int half = tid & 1;             // K half (16 bf16)
    const unsigned short* xsrc = xT + (size_t)(nbase + seg) * IN_ + half * 16;
    unsigned short* bdst = &Bs[0] + seg * BROW + half * 16;

    v8f acc[8];
    const v8f zero = {0.f, 0.f, 0.f, 0.f, 0.f, 0.f, 0.f, 0.f};
    #pragma unroll
    for (int nt = 0; nt < 8; ++nt) acc[nt] = zero;

    // prologue: stage tile kt=0, preload A fragment kt=0
    v16bf rcopy = *(const v16bf*)(xsrc);
    *(v16bf*)(bdst) = rcopy;
    v16bf a_cur = *(const v16bf*)(wg + (size_t)(mt * KT32) * 512);
    __syncthreads();

    #pragma unroll 1
    for (int kt = 0; kt < KT32; ++kt) {
        const unsigned short* bb = &Bs[0] + (kt & 1) * BUFE + col * BROW + (hi << 4);
        v16bf a_nxt;
        if (kt + 1 < KT32) {
            a_nxt = *(const v16bf*)(wg + (size_t)(mt * KT32 + kt + 1) * 512);
            rcopy = *(const v16bf*)(xsrc + (kt + 1) * 32);
        }
        // B fragments double-buffered from LDS: load nt+1 before WMMA nt
        v16bf b_cur = *(const v16bf*)(bb);
        #pragma unroll
        for (int nt = 0; nt < 8; ++nt) {
            v16bf b_nxt;
            if (nt < 7) b_nxt = *(const v16bf*)(bb + (nt + 1) * 16 * BROW);
            acc[nt] = wmma_bf16(a_cur, b_cur, acc[nt]);
            if (nt < 7) b_cur = b_nxt;
        }
        if (kt + 1 < KT32) {
            *(v16bf*)(&Bs[0] + ((kt + 1) & 1) * BUFE + seg * BROW + half * 16) = rcopy;
            a_cur = a_nxt;
            __syncthreads();
        }
    }

    float* pg = p + (size_t)g * HIDN * NTB;
    #pragma unroll
    for (int nt = 0; nt < 8; ++nt) {
        int n = nbase + nt * 16 + col;
        #pragma unroll
        for (int r = 0; r < 8; ++r) {
            int m = mt * 16 + (hi << 3) + r;        // C layout: VGPR r, lane>=16 → +8
            pg[(size_t)m * NTB + n] = acc[nt][r];
        }
    }
}

// ---------------------------------------------------------------------------
// One gate GEMM for the recurrence: acc[j] += Wg[rows mt0..mt0+3] @ bvec.
// Manual 2-stage software pipeline over kt (step 2 keeps buffer selection at
// compile time: no dynamic indexing of register arrays, no scratch). Both B
// fragments for the kt pair are loaded before the first WMMA group so the
// second group's ds_load is covered by 4 WMMAs.
// ---------------------------------------------------------------------------
static __device__ __forceinline__ void gate_gemm(const unsigned short* __restrict__ wg,
                                                 const unsigned short* bfragbase, // LDS: + col*HROW
                                                 int mt0, int lane, v8f acc[4])
{
    const int hi = lane >> 4;
    const unsigned short* wl = wg + (size_t)lane * 16;   // + tile*512
    v16bf a0[4], a1[4];
    #pragma unroll
    for (int j = 0; j < 4; ++j)
        a0[j] = *(const v16bf*)(wl + (size_t)((mt0 + j) * KT32) * 512);

    #pragma unroll 1
    for (int kt = 0; kt < KT32; kt += 2) {
        #pragma unroll
        for (int j = 0; j < 4; ++j)
            a1[j] = *(const v16bf*)(wl + (size_t)((mt0 + j) * KT32 + kt + 1) * 512);
        v16bf b0 = *(const v16bf*)(bfragbase + kt * 32 + (hi << 4));
        v16bf b1 = *(const v16bf*)(bfragbase + (kt + 1) * 32 + (hi << 4));
        #pragma unroll
        for (int j = 0; j < 4; ++j) acc[j] = wmma_bf16(a0[j], b0, acc[j]);

        if (kt + 2 < KT32) {
            #pragma unroll
            for (int j = 0; j < 4; ++j)
                a0[j] = *(const v16bf*)(wl + (size_t)((mt0 + j) * KT32 + kt + 2) * 512);
        }
        #pragma unroll
        for (int j = 0; j < 4; ++j) acc[j] = wmma_bf16(a1[j], b1, acc[j]);
    }
}

// ---------------------------------------------------------------------------
// Recurrence: 16 blocks, each owns 16 batch columns for all 128 steps.
// 512 threads = 16 waves; wave owns 4 M-tiles (64 hidden rows).
// h and r*h live in LDS as bf16 [b][k] (B-fragment layout: lane<16 K-low half,
// lane>=16 K-high half, contiguous per lane). Recurrent weights (6 MB bf16)
// stay L2-resident across all 128 steps.
// ---------------------------------------------------------------------------
__launch_bounds__(512, 1)
__global__ void gru_rec(const float* __restrict__ hx,            // [HIDN][BATCH]
                        const float* __restrict__ p,             // [3][HIDN][NTB]
                        const unsigned short* __restrict__ wH,   // packed [hh|hr|hn]
                        float* __restrict__ out)                 // [HIDN][NTB] + [HIDN][BATCH]
{
    __shared__ __attribute__((aligned(32))) unsigned short hT [16 * HROW];
    __shared__ __attribute__((aligned(32))) unsigned short rhT[16 * HROW];

    const int tid   = threadIdx.x;
    const int wave  = tid >> 5;
    const int lane  = tid & 31;
    const int col   = lane & 15;
    const int hi    = lane >> 4;
    const int bbase = blockIdx.x * 16;
    const int mt0   = wave * 4;

    // load initial hidden state into LDS (bf16, [b][k])
    for (int idx = tid; idx < 16 * HIDN; idx += 512) {
        int bb = idx >> 10;
        int k  = idx & (HIDN - 1);
        hT[bb * HROW + k] = f2bf(hx[(size_t)k * BATCH + bbase + bb]);
    }
    __syncthreads();

    const unsigned short* wHH = wH;
    const unsigned short* wHR = wH + 1024u * 1024u;
    const unsigned short* wHN = wH + 2u * 1024u * 1024u;
    const float* pI = p;
    const float* pR = p + (size_t)HIDN * NTB;
    const float* pN = p + 2 * (size_t)HIDN * NTB;

    const unsigned short* hfrag  = hT  + col * HROW;
    const unsigned short* rhfrag = rhT + col * HROW;

    for (int t = 0; t < TT; ++t) {
        // ---- gates i and r: acc = p + W @ h ----
        v8f accI[4], accR[4];
        #pragma unroll
        for (int j = 0; j < 4; ++j) {
            #pragma unroll
            for (int r = 0; r < 8; ++r) {
                size_t m = (size_t)((mt0 + j) * 16 + (hi << 3) + r);
                size_t o = m * NTB + (size_t)t * BATCH + bbase + col;
                accI[j][r] = pI[o];
                accR[j][r] = pR[o];
            }
        }
        gate_gemm(wHH, hfrag, mt0, lane, accI);
        gate_gemm(wHR, hfrag, mt0, lane, accR);

        // ---- i = sigmoid, r = sigmoid, publish r*h ----
        float hv[4][8];
        #pragma unroll
        for (int j = 0; j < 4; ++j) {
            #pragma unroll
            for (int r = 0; r < 8; ++r) {
                int m = (mt0 + j) * 16 + (hi << 3) + r;
                float h_old = bf2f(hT[col * HROW + m]);
                hv[j][r] = h_old;
                float ig = 1.0f / (1.0f + __expf(-accI[j][r]));
                accI[j][r] = ig;                                   // keep i-gate
                float rg = 1.0f / (1.0f + __expf(-accR[j][r]));
                rhT[col * HROW + m] = f2bf(rg * h_old);
            }
        }
        __syncthreads();   // r*h visible to all waves; all hT reads done

        // ---- gate n: acc = p_n + W_hn @ (r*h) ----
        v8f accN[4];
        #pragma unroll
        for (int j = 0; j < 4; ++j) {
            #pragma unroll
            for (int r = 0; r < 8; ++r) {
                size_t m = (size_t)((mt0 + j) * 16 + (hi << 3) + r);
                accN[j][r] = pN[m * NTB + (size_t)t * BATCH + bbase + col];
            }
        }
        gate_gemm(wHN, rhfrag, mt0, lane, accN);

        // ---- h_new = (1-i)*h + i*tanh(n); store output + update LDS h ----
        #pragma unroll
        for (int j = 0; j < 4; ++j) {
            #pragma unroll
            for (int r = 0; r < 8; ++r) {
                int m = (mt0 + j) * 16 + (hi << 3) + r;
                float ng   = tanhf(accN[j][r]);
                float ig   = accI[j][r];
                float hnew = (1.0f - ig) * hv[j][r] + ig * ng;
                out[(size_t)m * NTB + (size_t)t * BATCH + bbase + col] = hnew;
                hT[col * HROW + m] = f2bf(hnew);
                if (t == TT - 1)
                    out[(size_t)HIDN * NTB + (size_t)m * BATCH + bbase + col] = hnew;
            }
        }
        __syncthreads();   // h_new visible before next step's fragment reads
    }
}

// ---------------------------------------------------------------------------
extern "C" void kernel_launch(void* const* d_in, const int* in_sizes, int n_in,
                              void* d_out, int out_size, void* d_ws, size_t ws_size,
                              hipStream_t stream) {
    (void)in_sizes; (void)n_in; (void)out_size; (void)ws_size;

    const float* input = (const float*)d_in[0];   // [IN, T, B]
    const float* hx    = (const float*)d_in[1];   // [HID, B]
    const float* w_ih  = (const float*)d_in[2];
    const float* w_hh  = (const float*)d_in[3];
    const float* w_ir  = (const float*)d_in[4];
    const float* w_hr  = (const float*)d_in[5];
    const float* w_in  = (const float*)d_in[6];
    const float* w_hn  = (const float*)d_in[7];
    float* out = (float*)d_out;

    // workspace layout (hipMalloc base is 256B aligned; all sizes 256B multiples)
    char* ws = (char*)d_ws;
    size_t off = 0;
    unsigned short* xT   = (unsigned short*)(ws + off); off += (size_t)NTB * IN_ * 2;        // 64 MiB
    unsigned short* wpIn = (unsigned short*)(ws + off); off += (size_t)3 * 1024 * 1024 * 2;  // 6 MiB
    unsigned short* wpH  = (unsigned short*)(ws + off); off += (size_t)3 * 1024 * 1024 * 2;  // 6 MiB
    float*          pPrj = (float*)(ws + off);          off += (size_t)3 * HIDN * NTB * 4;   // 384 MiB

    // 1) pack weights into WMMA A-fragment layout (gate order: i, r, n)
    dim3 pgrid(4096), pblk(256);
    pack_w<<<pgrid, pblk, 0, stream>>>(w_ih, wpIn);
    pack_w<<<pgrid, pblk, 0, stream>>>(w_ir, wpIn + 1024u * 1024u);
    pack_w<<<pgrid, pblk, 0, stream>>>(w_in, wpIn + 2u * 1024u * 1024u);
    pack_w<<<pgrid, pblk, 0, stream>>>(w_hh, wpH);
    pack_w<<<pgrid, pblk, 0, stream>>>(w_hr, wpH + 1024u * 1024u);
    pack_w<<<pgrid, pblk, 0, stream>>>(w_hn, wpH + 2u * 1024u * 1024u);

    // 2) transpose-convert x to bf16 [N][K]
    transpose_x<<<dim3(NTB / 32, IN_ / 32), 256, 0, stream>>>(input, xT);

    // 3) input projections (big parallel WMMA GEMM, LDS-staged B tiles)
    gemm_proj<<<dim3(NTB / 128, HIDN / 128, 3), 256, 0, stream>>>(wpIn, xT, pPrj);

    // 4) serial recurrence, batch-sliced across 16 persistent blocks
    gru_rec<<<dim3(BATCH / 16), 512, 0, stream>>>(hx, pPrj, wpH, out);
}
